// MT3Decoder_45071386804661
// MI455X (gfx1250) — matrix operand order repaired and checked by reading, not compiled, not run on hardware
//
#include <hip/hip_runtime.h>

// ---------------------------------------------------------------------------
// MT3-style T5 decoder forward on gfx1250 (MI455X): bf16 WMMA everywhere.
// B=4, LT=1024, LS=2048, D=1024, H=16, dh=64, DFF=4096, NL=8
// All WMMA fragments are loaded as two contiguous b128 chunks per lane:
//   A 16x32 : lane(m,g) -> row m, K[16g..16g+15]
//   B 32x16 : lane(n,g) -> col n, K[16g..16g+15]   (needs K-contiguous per n)
// so weights are pre-transposed to [N][K] bf16 and V to [B,H,dh,L] bf16.
// ---------------------------------------------------------------------------

typedef __attribute__((ext_vector_type(16))) __bf16 v16bf;
typedef __attribute__((ext_vector_type(8)))  float  v8f;

#define DEV __device__ __forceinline__

constexpr int B_  = 4;
constexpr int LT  = 1024;
constexpr int LS  = 2048;
constexpr int D_  = 1024;
constexpr int H_  = 16;
constexpr int DH  = 64;
constexpr int DFF = 4096;
constexpr int NL  = 8;

DEV __bf16 f2bf(float f) {
  union { float f; unsigned u; } x; x.f = f;
  unsigned r = (x.u + 0x7fffu + ((x.u >> 16) & 1u)) >> 16;   // RNE
  union { unsigned short us; __bf16 b; } y; y.us = (unsigned short)r;
  return y.b;
}

DEV v8f wmma_bf16(v16bf a, v16bf b, v8f c) {
  return __builtin_amdgcn_wmma_f32_16x16x32_bf16(false, a, false, b, (short)0, c,
                                                 false, false);
}

union FragCvt { uint4 q[2]; v16bf v; };

// Two 16B chunks -> one 16xbf16 fragment register set
DEV v16bf frag2(const void* p0, const void* p1) {
  FragCvt u;
  u.q[0] = *(const uint4*)p0;
  u.q[1] = *(const uint4*)p1;
  return u.v;
}

// A fragment 16x32 (MxK) from row-major [m][k], leading dim ld (elements).
DEV v16bf load_a_frag(const __bf16* base, int ld, int lane) {
  int m = lane & 15, g = lane >> 4;
  const __bf16* p = base + (size_t)m * ld + 16 * g;
  return frag2(p, p + 8);
}

// B fragment 32x16 (KxN) from K-contiguous storage: bt[n][k], leading dim ld.
DEV v16bf load_bt_frag(const __bf16* bt, size_t ld, int n_base, int k_base, int lane) {
  int n = lane & 15, g = lane >> 4;
  const __bf16* p = bt + (size_t)(n_base + n) * ld + k_base + 16 * g;
  return frag2(p, p + 8);
}

// gfx1250 async global->LDS staging (CDNA5 ISA: VDST=LDS offset, VADDR=64b)
DEV void async_lds_b128(unsigned ldsoff, const void* gptr) {
  asm volatile("global_load_async_to_lds_b128 %0, %1, off"
               :: "v"(ldsoff), "v"(gptr) : "memory");
}
DEV void wait_async0() { asm volatile("s_wait_asynccnt 0x0" ::: "memory"); }

// ---------------------------------------------------------------------------
// Elementwise helpers
// ---------------------------------------------------------------------------
__global__ void cvt_f32_bf16(const float* __restrict__ s, __bf16* __restrict__ d, int n) {
  int i = blockIdx.x * blockDim.x + threadIdx.x;
  if (i < n) d[i] = f2bf(s[i]);
}

__global__ void copy_f32(const float* __restrict__ s, float* __restrict__ d, int n) {
  int i = blockIdx.x * blockDim.x + threadIdx.x;
  if (i < n) d[i] = s[i];
}

// W [K][N] fp32 -> WT [N][K] bf16, 32x32 LDS tile transpose. grid(K/32, N/32)
__global__ __launch_bounds__(256)
void cvt_transpose(const float* __restrict__ w, __bf16* __restrict__ wt,
                   int Kd, int Nd) {
  __shared__ float t[32][33];
  int k0 = blockIdx.x * 32, n0 = blockIdx.y * 32;
  int tx = threadIdx.x & 31, ty = threadIdx.x >> 5;
  for (int i = ty; i < 32; i += 8)
    t[i][tx] = w[(size_t)(k0 + i) * Nd + n0 + tx];
  __syncthreads();
  for (int i = ty; i < 32; i += 8)
    wt[(size_t)(n0 + i) * Kd + k0 + tx] = f2bf(t[tx][i]);
}

// attn out [B,H,L,64] fp32 -> merged [B*L, D] bf16
__global__ void merge_heads(const float* __restrict__ in, __bf16* __restrict__ out, int L) {
  int i = blockIdx.x * blockDim.x + threadIdx.x;
  if (i >= B_ * H_ * L * DH) return;
  int dd = i & (DH - 1);
  int t = i >> 6;
  int ll = t % L; t /= L;
  int hh = t & (H_ - 1);
  int bb = t >> 4;
  out[((size_t)bb * L + ll) * D_ + hh * DH + dd] = f2bf(in[i]);
}

// row LayerNorm over D=1024, output bf16 (GEMM input)
__global__ __launch_bounds__(256)
void ln_kernel(const float* __restrict__ x, const float* __restrict__ g,
               const float* __restrict__ b, __bf16* __restrict__ out) {
  int row = blockIdx.x;
  const float* xr = x + (size_t)row * D_;
  float s1 = 0.f, s2 = 0.f;
  for (int i = threadIdx.x; i < D_; i += 256) { float v = xr[i]; s1 += v; s2 += v * v; }
  __shared__ float r1[256], r2[256];
  r1[threadIdx.x] = s1; r2[threadIdx.x] = s2;
  __syncthreads();
  for (int off = 128; off > 0; off >>= 1) {
    if (threadIdx.x < off) { r1[threadIdx.x] += r1[threadIdx.x + off];
                             r2[threadIdx.x] += r2[threadIdx.x + off]; }
    __syncthreads();
  }
  float mean = r1[0] * (1.0f / D_);
  float var  = r2[0] * (1.0f / D_) - mean * mean;
  float rs = rsqrtf(var + 1e-5f);
  for (int i = threadIdx.x; i < D_; i += 256)
    out[(size_t)row * D_ + i] = f2bf((xr[i] - mean) * rs * g[i] + b[i]);
}

// ---------------------------------------------------------------------------
// GEMM: C[M,N] = A[M,K](bf16) x Bt[N,K](bf16, pre-transposed), fp32 acc WMMA.
// Block 256 = 8 waves; tile 128x64; wave = one 32x32 accumulator.
// A tile staged to LDS via async b128 loads; B fragments straight from global.
// MODE 0: out fp32 = acc + bias + resid   (resid may alias out)
// MODE 1: out bf16 = relu(acc + bias)
// MODE 2: out bf16 head-split [B,H,L,64], value (acc+bias)*scale
// MODE 3: out bf16 head-split transposed [B,H,64,L] (for V)
// ---------------------------------------------------------------------------
template <int MODE>
__global__ __launch_bounds__(256)
void gemm_bf16_kernel(const __bf16* __restrict__ A, const __bf16* __restrict__ Bt,
                      const float* __restrict__ bias, void* __restrict__ out,
                      const float* __restrict__ resid, int M, int N, int K,
                      int L, float scale) {
  __shared__ __align__(16) __bf16 sA[128 * 40];   // 128 rows x 32 K, pad to 40
  const int tid = threadIdx.x, lane = tid & 31, wv = tid >> 5;
  const int wm = wv >> 1, wn = wv & 1;
  const int m0 = blockIdx.y * 128, n0 = blockIdx.x * 64;

  v8f a00 = {}, a01 = {}, a10 = {}, a11 = {};

  for (int kk = 0; kk < K; kk += 32) {
    // stage A tile (128x32 bf16 = 8KB) with async global->LDS, 2x16B/thread
#pragma unroll
    for (int it = 0; it < 2; ++it) {
      int c = tid + it * 256;
      int row = c >> 2, col = (c & 3) * 8;
      unsigned loff = (unsigned)(uintptr_t)&sA[row * 40 + col];
      async_lds_b128(loff, A + (size_t)(m0 + row) * K + kk + col);
    }
    if (kk + 32 < K)
      __builtin_prefetch(Bt + (size_t)(n0 + (tid & 63)) * K + kk + 32, 0, 1);
    wait_async0();
    __syncthreads();

    v16bf fa0 = load_a_frag(&sA[(wm * 32 + 0)  * 40], 40, lane);
    v16bf fa1 = load_a_frag(&sA[(wm * 32 + 16) * 40], 40, lane);
    v16bf fb0 = load_bt_frag(Bt, K, n0 + wn * 32 + 0,  kk, lane);
    v16bf fb1 = load_bt_frag(Bt, K, n0 + wn * 32 + 16, kk, lane);
    a00 = wmma_bf16(fa0, fb0, a00);
    a01 = wmma_bf16(fa0, fb1, a01);
    a10 = wmma_bf16(fa1, fb0, a10);
    a11 = wmma_bf16(fa1, fb1, a11);
    __syncthreads();
  }

  const int nl = lane & 15, g = lane >> 4;
  v8f accs[2][2] = {{a00, a01}, {a10, a11}};
#pragma unroll
  for (int i = 0; i < 2; ++i)
#pragma unroll
    for (int j = 0; j < 2; ++j)
#pragma unroll
      for (int r = 0; r < 8; ++r) {
        int row = m0 + wm * 32 + i * 16 + r + 8 * g;
        int col = n0 + wn * 32 + j * 16 + nl;
        float v = accs[i][j][r] + bias[col];
        if (MODE == 0) {
          float* o = (float*)out;
          o[(size_t)row * N + col] = v + resid[(size_t)row * N + col];
        } else if (MODE == 1) {
          __bf16* o = (__bf16*)out;
          o[(size_t)row * N + col] = f2bf(fmaxf(v, 0.f));
        } else {
          __bf16* o = (__bf16*)out;
          int hd = col >> 6, dd = col & 63;
          int bb = row / L, ll = row % L;
          if (MODE == 2)
            o[(((size_t)bb * H_ + hd) * L + ll) * DH + dd] = f2bf(v * scale);
          else
            o[(((size_t)bb * H_ + hd) * DH + dd) * L + ll] = f2bf(v);
        }
      }
}

// ---------------------------------------------------------------------------
// Flash attention (online softmax). One wave per 16-row q-tile, 4 waves/block.
// Q pre-scaled by 1/sqrt(dh). causal!=0: T5 rel-pos bias + causal mask.
// Q,K: bf16 [B,H,L,64]; V: bf16 [B,H,64,L] (transposed); O: fp32 [B,H,Lq,64].
// ---------------------------------------------------------------------------
__global__ __launch_bounds__(128)
void attn_kernel(const __bf16* __restrict__ Q, const __bf16* __restrict__ K,
                 const __bf16* __restrict__ V, float* __restrict__ O,
                 const float* __restrict__ rel, int Lq, int Lk, int causal) {
  __shared__ __align__(16) __bf16 smemP[4][16 * 32];  // wave-private P staging
  const int lane = threadIdx.x & 31;
  const int wv = threadIdx.x >> 5;
  const int q0 = blockIdx.x * 64 + wv * 16;
  const int h = blockIdx.y, b = blockIdx.z;
  const __bf16* Qb = Q + ((size_t)(b * H_ + h) * Lq) * DH;
  const __bf16* Kb = K + ((size_t)(b * H_ + h) * Lk) * DH;
  const __bf16* Vt = V + ((size_t)(b * H_ + h) * DH) * Lk;

  v16bf qf0 = load_a_frag(Qb + (size_t)q0 * DH, DH, lane);       // dh 0..31
  v16bf qf1 = load_a_frag(Qb + (size_t)q0 * DH + 32, DH, lane);  // dh 32..63

  v8f o0 = {}, o1 = {}, o2 = {}, o3 = {};
  float Mr[8], Sr[8];
#pragma unroll
  for (int r = 0; r < 8; ++r) { Mr[r] = -1e30f; Sr[r] = 0.f; }

  const int nl = lane & 15, g = lane >> 4;
  const float invlog = 16.0f / __logf(8.0f);  // (NB-16)/log(MAXD/16)
  const int kend = causal ? (q0 + 16) : Lk;

  for (int kc = 0; kc < kend; kc += 32) {
    v8f z = {};
    // S = Q K^T : B[k][n] = K[key0+n][dh0+k], contiguous along dh
    v8f s0 = wmma_bf16(qf0, load_bt_frag(Kb, DH, kc, 0, lane), z);
    s0 = wmma_bf16(qf1, load_bt_frag(Kb, DH, kc, 32, lane), s0);
    v8f s1 = wmma_bf16(qf0, load_bt_frag(Kb, DH, kc + 16, 0, lane), z);
    s1 = wmma_bf16(qf1, load_bt_frag(Kb, DH, kc + 16, 32, lane), s1);

    float sv[2][8];
#pragma unroll
    for (int r = 0; r < 8; ++r) { sv[0][r] = s0[r]; sv[1][r] = s1[r]; }

    if (causal) {
#pragma unroll
      for (int sub = 0; sub < 2; ++sub) {
        int key = kc + sub * 16 + nl;
#pragma unroll
        for (int r = 0; r < 8; ++r) {
          int qr = q0 + r + 8 * g;
          int rp = key - qr;
          if (rp > 0) sv[sub][r] = -1e9f;
          else {
            int n = -rp;  // T5 unidirectional bucket
            int bucket;
            if (n < 16) bucket = n;
            else {
              int lg = 16 + (int)(__logf((float)n * 0.0625f) * invlog);
              bucket = lg < 31 ? lg : 31;
            }
            sv[sub][r] += rel[bucket * H_ + h];
          }
        }
      }
    }

    float corr[8];
#pragma unroll
    for (int r = 0; r < 8; ++r) {
      float v = fmaxf(sv[0][r], sv[1][r]);
      for (int off = 8; off > 0; off >>= 1) v = fmaxf(v, __shfl_xor(v, off, 16));
      float nm = fmaxf(Mr[r], v);
      corr[r] = __expf(Mr[r] - nm);
      float p0 = __expf(sv[0][r] - nm);
      float p1 = __expf(sv[1][r] - nm);
      sv[0][r] = p0; sv[1][r] = p1;
      float rs = p0 + p1;
      for (int off = 8; off > 0; off >>= 1) rs += __shfl_xor(rs, off, 16);
      Sr[r] = Sr[r] * corr[r] + rs;
      Mr[r] = nm;
    }
#pragma unroll
    for (int r = 0; r < 8; ++r) {
      o0[r] *= corr[r]; o1[r] *= corr[r]; o2[r] *= corr[r]; o3[r] *= corr[r];
    }

    // C-layout -> A-fragment via wave-private LDS (same-wave DS ops in order)
    __bf16* pw = smemP[wv];
#pragma unroll
    for (int sub = 0; sub < 2; ++sub)
#pragma unroll
      for (int r = 0; r < 8; ++r)
        pw[(r + 8 * g) * 32 + sub * 16 + nl] = f2bf(sv[sub][r]);
    v16bf pf = load_a_frag(pw, 32, lane);

    // O += P V : B[k][n] = V[key0+k][dh n] = Vt[n][key0+k], contiguous keys
    o0 = wmma_bf16(pf, load_bt_frag(Vt, Lk, 0,  kc, lane), o0);
    o1 = wmma_bf16(pf, load_bt_frag(Vt, Lk, 16, kc, lane), o1);
    o2 = wmma_bf16(pf, load_bt_frag(Vt, Lk, 32, kc, lane), o2);
    o3 = wmma_bf16(pf, load_bt_frag(Vt, Lk, 48, kc, lane), o3);
  }

  float* Ob = O + ((size_t)(b * H_ + h) * Lq + q0) * DH;
#pragma unroll
  for (int r = 0; r < 8; ++r) {
    float inv = 1.0f / Sr[r];
    int row = r + 8 * g;
    Ob[(size_t)row * DH + 0  + nl] = o0[r] * inv;
    Ob[(size_t)row * DH + 16 + nl] = o1[r] * inv;
    Ob[(size_t)row * DH + 32 + nl] = o2[r] * inv;
    Ob[(size_t)row * DH + 48 + nl] = o3[r] * inv;
  }
}

// ---------------------------------------------------------------------------
// Orchestration
// ---------------------------------------------------------------------------
extern "C" void kernel_launch(void* const* d_in, const int* in_sizes, int n_in,
                              void* d_out, int out_size, void* d_ws, size_t ws_size,
                              hipStream_t stream) {
  const float* tgt = (const float*)d_in[0];
  const float* mem = (const float*)d_in[1];
  const float* sa_wq = (const float*)d_in[2];
  const float* sa_bq = (const float*)d_in[3];
  const float* sa_wk = (const float*)d_in[4];
  const float* sa_bk = (const float*)d_in[5];
  const float* sa_wv = (const float*)d_in[6];
  const float* sa_bv = (const float*)d_in[7];
  const float* sa_wo = (const float*)d_in[8];
  const float* sa_bo = (const float*)d_in[9];
  const float* ca_wq = (const float*)d_in[10];
  const float* ca_bq = (const float*)d_in[11];
  const float* ca_wk = (const float*)d_in[12];
  const float* ca_bk = (const float*)d_in[13];
  const float* ca_wv = (const float*)d_in[14];
  const float* ca_bv = (const float*)d_in[15];
  const float* ca_wo = (const float*)d_in[16];
  const float* ca_bo = (const float*)d_in[17];
  const float* ffn_w1 = (const float*)d_in[18];
  const float* ffn_b1 = (const float*)d_in[19];
  const float* ffn_w2 = (const float*)d_in[20];
  const float* ffn_b2 = (const float*)d_in[21];
  const float* ln1_g = (const float*)d_in[22];
  const float* ln1_b = (const float*)d_in[23];
  const float* ln2_g = (const float*)d_in[24];
  const float* ln2_b = (const float*)d_in[25];
  const float* ln3_g = (const float*)d_in[26];
  const float* ln3_b = (const float*)d_in[27];
  const float* rel   = (const float*)d_in[28];

  char* ws = (char*)d_ws;
  __bf16* hbuf = (__bf16*)(ws + (0ull   << 20));  //  8 MB  [B*LT, D] bf16
  __bf16* memb = (__bf16*)(ws + (8ull   << 20));  // 16 MB  [B*LS, D] bf16
  __bf16* qh   = (__bf16*)(ws + (24ull  << 20));  //  8 MB  [B,H,LT,64]
  __bf16* kh   = (__bf16*)(ws + (32ull  << 20));  // 16 MB  [B,H,LS,64] max
  __bf16* vh   = (__bf16*)(ws + (48ull  << 20));  // 16 MB  [B,H,64,LS] max
  float*  aof  = (float*) (ws + (64ull  << 20));  // 16 MB  [B,H,LT,64] fp32
  __bf16* amrg = (__bf16*)(ws + (80ull  << 20));  //  8 MB  [B*LT, D]
  __bf16* ffm  = (__bf16*)(ws + (88ull  << 20));  // 32 MB  [B*LT, DFF]
  __bf16* wsc  = (__bf16*)(ws + (120ull << 20));  //  8 MB  WT scratch [N][K]
  float* x = (float*)d_out;

  const int nTok = B_ * LT;   // 4096
  const int nMem = B_ * LS;   // 8192

  copy_f32<<<(B_ * LT * D_) / 256, 256, 0, stream>>>(tgt, x, B_ * LT * D_);
  cvt_f32_bf16<<<(B_ * LS * D_) / 256, 256, 0, stream>>>(mem, memb, B_ * LS * D_);

  // convert + transpose a weight [K][N] fp32 -> wsc [N][K] bf16
  auto cvtw = [&](const float* w, int Kd, int Nd) {
    cvt_transpose<<<dim3(Kd / 32, Nd / 32), 256, 0, stream>>>(w, wsc, Kd, Nd);
  };

  dim3 gProj(D_ / 64, nTok / 128);   // (16, 32)
  dim3 gProjM(D_ / 64, nMem / 128);  // (16, 64)
  dim3 gFf1(DFF / 64, nTok / 128);   // (64, 32)
  dim3 gAttn(LT / 64, H_, B_);

  for (int l = 0; l < NL; ++l) {
    size_t oDD = (size_t)l * D_ * D_;
    size_t oD  = (size_t)l * D_;
    size_t oF1 = (size_t)l * D_ * DFF;
    size_t oFb = (size_t)l * DFF;

    // ---- self attention ----
    ln_kernel<<<nTok, 256, 0, stream>>>(x, ln1_g + oD, ln1_b + oD, hbuf);
    cvtw(sa_wq + oDD, D_, D_);
    gemm_bf16_kernel<2><<<gProj, 256, 0, stream>>>(hbuf, wsc, sa_bq + oD, qh, nullptr,
                                                   nTok, D_, D_, LT, 0.125f);
    cvtw(sa_wk + oDD, D_, D_);
    gemm_bf16_kernel<2><<<gProj, 256, 0, stream>>>(hbuf, wsc, sa_bk + oD, kh, nullptr,
                                                   nTok, D_, D_, LT, 1.0f);
    cvtw(sa_wv + oDD, D_, D_);
    gemm_bf16_kernel<3><<<gProj, 256, 0, stream>>>(hbuf, wsc, sa_bv + oD, vh, nullptr,
                                                   nTok, D_, D_, LT, 1.0f);
    attn_kernel<<<gAttn, 128, 0, stream>>>(qh, kh, vh, aof, rel, LT, LT, 1);
    merge_heads<<<(B_ * H_ * LT * DH) / 256, 256, 0, stream>>>(aof, amrg, LT);
    cvtw(sa_wo + oDD, D_, D_);
    gemm_bf16_kernel<0><<<gProj, 256, 0, stream>>>(amrg, wsc, sa_bo + oD, x, x,
                                                   nTok, D_, D_, 0, 1.0f);

    // ---- cross attention ----
    ln_kernel<<<nTok, 256, 0, stream>>>(x, ln2_g + oD, ln2_b + oD, hbuf);
    cvtw(ca_wq + oDD, D_, D_);
    gemm_bf16_kernel<2><<<gProj, 256, 0, stream>>>(hbuf, wsc, ca_bq + oD, qh, nullptr,
                                                   nTok, D_, D_, LT, 0.125f);
    cvtw(ca_wk + oDD, D_, D_);
    gemm_bf16_kernel<2><<<gProjM, 256, 0, stream>>>(memb, wsc, ca_bk + oD, kh, nullptr,
                                                    nMem, D_, D_, LS, 1.0f);
    cvtw(ca_wv + oDD, D_, D_);
    gemm_bf16_kernel<3><<<gProjM, 256, 0, stream>>>(memb, wsc, ca_bv + oD, vh, nullptr,
                                                    nMem, D_, D_, LS, 1.0f);
    attn_kernel<<<gAttn, 128, 0, stream>>>(qh, kh, vh, aof, rel, LT, LS, 0);
    merge_heads<<<(B_ * H_ * LT * DH) / 256, 256, 0, stream>>>(aof, amrg, LT);
    cvtw(ca_wo + oDD, D_, D_);
    gemm_bf16_kernel<0><<<gProj, 256, 0, stream>>>(amrg, wsc, ca_bo + oD, x, x,
                                                   nTok, D_, D_, 0, 1.0f);

    // ---- FFN ----
    ln_kernel<<<nTok, 256, 0, stream>>>(x, ln3_g + oD, ln3_b + oD, hbuf);
    cvtw(ffn_w1 + oF1, D_, DFF);
    gemm_bf16_kernel<1><<<gFf1, 256, 0, stream>>>(hbuf, wsc, ffn_b1 + oFb, ffm, nullptr,
                                                  nTok, DFF, D_, 0, 1.0f);
    cvtw(ffn_w2 + oF1, DFF, D_);
    gemm_bf16_kernel<0><<<gProj, 256, 0, stream>>>(ffm, wsc, ffn_b2 + oD, x, x,
                                                   nTok, D_, DFF, 0, 1.0f);
  }
  (void)in_sizes; (void)n_in; (void)out_size; (void)ws_size;
}
